// UGRNNModel_65489661329496
// MI455X (gfx1250) — matrix operand order repaired and checked by reading, not compiled, not run on hardware
//
#include <hip/hip_runtime.h>
#include <math.h>

// ---------------- CDNA5 WMMA types ----------------
typedef __attribute__((ext_vector_type(16))) _Float16 v16h;
typedef __attribute__((ext_vector_type(8)))  float    v8f;

union V16 { v16h v; uint4 q[2]; };

// ---------------- problem constants ----------------
#define BDIM 64
#define NDIM 4096
#define KDIM 32
#define UDIM 16
#define STEPS 8
#define ROWS (BDIM*NDIM)            // 262144 graph nodes
#define TILES (ROWS/16)             // 16384 WMMA row-tiles
#define KTOT (NDIM*UDIM)            // 65536 = K of big GEMM
#define KC_TOT (KTOT/32)            // 2048 K-chunks of 32
#define KSLICES 64
#define KC_PER_SLICE (KC_TOT/KSLICES) // 32

// ---------------- workspace layout (bytes) ----------------
#define WS_RW_OFF   0u                         // 4 blobs x 512 halfs = 4 KB
#define WS_H_OFF    4096u                      // H f16: 64*65536 halfs = 8 MB
#define WS_W1_OFF   (4096u + 8388608u)         // W1 f16 blob: 32 MB
#define WS_PART_OFF (WS_W1_OFF + 33554432u)    // split-K partials: 4 MB f32
#define WS_OUT1_OFF (WS_PART_OFF + 4194304u)   // out1: 64*256 f32 = 64 KB
// total ~44.1 MB of d_ws

// ---------------- branchless HW transcendentals ----------------
// Keep EXEC all-1s (WMMA requirement) and use single-op v_exp_f32 / v_rcp_f32
// (TRANS ops co-execute with XDL WMMA on CDNA5).
#define LOG2E 1.4426950408889634f
__device__ __forceinline__ float fast_sigmoid(float x) {
  float e = __builtin_amdgcn_exp2f(-LOG2E * x);        // exp(-x)
  return __builtin_amdgcn_rcpf(1.0f + e);
}
__device__ __forceinline__ float fast_tanh(float x) {
  float e = __builtin_amdgcn_exp2f(-2.0f * LOG2E * x); // exp(-2x)
  return 2.0f * __builtin_amdgcn_rcpf(1.0f + e) - 1.0f;
}
__device__ __forceinline__ float fast_exp(float x) {
  return __builtin_amdgcn_exp2f(LOG2E * x);
}

// ======================================================================
// Prep 1: recurrent weight blobs in WMMA B-matrix (32x16 f16) lane layout.
// Blob q, lane l, element j:  K = (j>=8)*16 + (l>=16)*8 + (j&7), N = l&15
//   q=0: Wn (32x16)   q=1: [Wg;Ug]   q=2: [Wc;Uc]   q=3: [We;0]
// ======================================================================
__global__ void prep_rw(const float* __restrict__ Wn, const float* __restrict__ Wg,
                        const float* __restrict__ Ug, const float* __restrict__ Wc,
                        const float* __restrict__ Uc, const float* __restrict__ We,
                        _Float16* __restrict__ rw) {
  int tid = blockIdx.x * blockDim.x + threadIdx.x;     // 0..2047
  if (tid >= 4 * 512) return;
  int j = tid & 15, l = (tid >> 4) & 31, q = tid >> 9;
  int K = ((j >> 3) << 4) + ((l >> 4) << 3) + (j & 7), N = l & 15;
  float v;
  if (q == 0)      v = Wn[K * UDIM + N];
  else if (q == 1) v = (K < 16) ? Wg[K * UDIM + N] : Ug[(K - 16) * UDIM + N];
  else if (q == 2) v = (K < 16) ? Wc[K * UDIM + N] : Uc[(K - 16) * UDIM + N];
  else             v = (K < 16) ? We[K * UDIM + N] : 0.0f;
  rw[tid] = (_Float16)v;
}

// ======================================================================
// Prep 2: swizzle W1 (65536x256 f32, row-major) into f16 blob in B-lane
// layout: blob half index = (kc*16 + nt)*512 + l*16 + j.
// Coalesced f32 reads, scattered b16 writes (absorbed by L2).
// ======================================================================
__global__ void prep_w1(const float* __restrict__ W1, _Float16* __restrict__ w1b) {
  int tid = blockIdx.x * 256 + threadIdx.x;            // 0..16777215
  int k = tid >> 8, col = tid & 255;
  int kc = k >> 5, klocal = k & 31;
  int lh = (klocal >> 3) & 1, j = ((klocal >> 4) << 3) + (klocal & 7);
  int nt = col >> 4, l = (lh << 4) + (col & 15);
  int dst = ((kc * 16 + nt) << 9) + (l << 4) + j;
  w1b[dst] = (_Float16)W1[tid];
}

// ======================================================================
// Recurrent UGRNN: one wave owns 16 nodes. Per step, the step-input
// [x|h] (16x32 f16) is rebuilt in LDS (A-matrix layout), then 3 WMMAs
// against [Wg;Ug], [Wc;Uc], [We;0]. m = neigh@Wn is one K=32 WMMA.
// Fully unrolled: 25 static v_wmma_f32_16x16x32_f16, branchless epilogue.
// ======================================================================
__global__ __launch_bounds__(256) void ugrnn_steps(
    const float* __restrict__ inp, const _Float16* __restrict__ rw,
    const float* __restrict__ bg, const float* __restrict__ bc,
    const float* __restrict__ be, _Float16* __restrict__ H) {
  __shared__ __align__(16) _Float16 smem[8 * 512];     // 1 KB A-region per wave
  const int tid = threadIdx.x;
  const int wave = tid >> 5, l = tid & 31;
  const int lm = l & 15, lh = l >> 4;
  const int tile = blockIdx.x * 8 + wave;
  const long rowBase = (long)tile * 16;

  // B-layout weight blobs -> registers (2 x b128 each; hot in L2)
  V16 wn, wgu, wcu, we0;
  {
    const uint4* p = (const uint4*)rw;                 // 64 uint4 per blob
    int o = l * 2;
    wn.q[0]  = p[o];       wn.q[1]  = p[o + 1];
    wgu.q[0] = p[64 + o];  wgu.q[1] = p[64 + o + 1];
    wcu.q[0] = p[128 + o]; wcu.q[1] = p[128 + o + 1];
    we0.q[0] = p[192 + o]; we0.q[1] = p[192 + o + 1];
  }

  // A-matrix for neigh (16x32): lane holds row M=lm, K chunks lh*8.. / 16+lh*8..
  V16 an;
  {
    const float* rp = inp + (rowBase + lm) * 34 + 1;   // neigh features at cols 1..32
#pragma unroll
    for (int j = 0; j < 8; j++) an.v[j]     = (_Float16)rp[lh * 8 + j];
#pragma unroll
    for (int j = 0; j < 8; j++) an.v[8 + j] = (_Float16)rp[16 + lh * 8 + j];
  }
  v8f zero = {};
  v8f m = __builtin_amdgcn_wmma_f32_16x16x32_f16(false, an.v, false, wn.v,
                                                 (short)0, zero, false, false);

  // h/e start as broadcast scalars per node; kept in C-layout (lane: M=r+8*lh, N=lm)
  float h[8], e[8];
#pragma unroll
  for (int r = 0; r < 8; r++) {
    long rr = rowBase + r + 8 * lh;
    h[r] = inp[rr * 34];
    e[r] = inp[rr * 34 + 33];
  }
  const float bgl = bg[lm], bcl = bc[lm], bel = be[lm];

  _Float16* areg = smem + wave * 512;                  // [M][32] halfs: x in k<16, h in k>=16
  const uint4* lp = (const uint4*)areg;

#pragma unroll
  for (int s = 0; s < STEPS; s++) {
    // C-layout -> A-layout transpose through LDS
#pragma unroll
    for (int r = 0; r < 8; r++) {
      int M = r + 8 * lh;
      areg[M * 32 + lm]      = (_Float16)(m[r] + e[r]); // x = m + e
      areg[M * 32 + 16 + lm] = (_Float16)h[r];
    }
    asm volatile("s_wait_dscnt 0x0" ::: "memory");     // CDNA5 split DS counter wait
    V16 A;
    A.q[0] = lp[lm * 4 + lh];                          // K = lh*8 .. +7
    A.q[1] = lp[lm * 4 + 2 + lh];                      // K = 16+lh*8 .. +7

    v8f gacc = __builtin_amdgcn_wmma_f32_16x16x32_f16(false, A.v, false, wgu.v,
                                                      (short)0, zero, false, false);
    v8f cacc = __builtin_amdgcn_wmma_f32_16x16x32_f16(false, A.v, false, wcu.v,
                                                      (short)0, zero, false, false);
    v8f eacc = __builtin_amdgcn_wmma_f32_16x16x32_f16(false, A.v, false, we0.v,
                                                      (short)0, zero, false, false);
#pragma unroll
    for (int r = 0; r < 8; r++) {
      float g = fast_sigmoid(gacc[r] + bgl);
      float c = fast_tanh(cacc[r] + bcl);
      h[r] = g * h[r] + (1.0f - g) * c;
      e[r] = fast_tanh(eacc[r] + bel);
    }
  }

  // H[b*65536 + n*16 + u] = h  (row-major flat, matches stage-2 A layout)
#pragma unroll
  for (int r = 0; r < 8; r++) {
    long rr = rowBase + r + 8 * lh;
    H[rr * 16 + lm] = (_Float16)h[r];
  }
}

// ======================================================================
// Big GEMM: out1_partial = H(64x65536,f16) @ W1blob(f16), split-K.
// wave = (slice, mt, nt); 32 K-steps of 32; per step: 4 x b128 loads,
// 1 prefetch, 1 v_wmma_f32_16x16x32_f16. Deterministic partials (no atomics).
// ======================================================================
__global__ __launch_bounds__(256) void gemm1(const _Float16* __restrict__ H,
                                             const _Float16* __restrict__ w1b,
                                             float* __restrict__ part) {
  const int tid = threadIdx.x;
  const int wave = tid >> 5, l = tid & 31;
  const int lm = l & 15, lh = l >> 4;
  const int wid = blockIdx.x * 8 + wave;
  const int slice = wid >> 6;
  const int t = wid & 63;
  const int mt = t >> 4, nt = t & 15;

  const uint4* Ap = (const uint4*)(H + (long)(mt * 16 + lm) * KTOT); // 8192 uint4/row
  v8f acc = {};
  int kc0 = slice * KC_PER_SLICE;
#pragma unroll 4
  for (int s = 0; s < KC_PER_SLICE; s++) {
    int kc = kc0 + s;
    V16 A, Bv;
    A.q[0] = Ap[kc * 4 + lh];
    A.q[1] = Ap[kc * 4 + 2 + lh];
    long bo = ((long)(kc * 16 + nt) << 9) + l * 16;    // half offset into blob
    const uint4* bq = (const uint4*)(w1b + bo);
    Bv.q[0] = bq[0];
    Bv.q[1] = bq[1];
    if (s + 1 < KC_PER_SLICE)
      __builtin_prefetch(w1b + bo + 8192, 0, 1);       // next kc's B tile -> global_prefetch
    acc = __builtin_amdgcn_wmma_f32_16x16x32_f16(false, A.v, false, Bv.v,
                                                 (short)0, acc, false, false);
  }
  float* pp = part + (long)slice * 16384 + (mt * 16) * 256 + nt * 16 + lm;
#pragma unroll
  for (int r = 0; r < 8; r++) pp[(r + 8 * lh) * 256] = acc[r];
}

// ======================================================================
// Split-K reduction + bias + relu -> out1 (64x256 f32)
// ======================================================================
__global__ void reduce1(const float* __restrict__ part, const float* __restrict__ b1,
                        float* __restrict__ out1) {
  int i = blockIdx.x * 256 + threadIdx.x;              // 0..16383
  float s = 0.0f;
#pragma unroll 8
  for (int sl = 0; sl < KSLICES; sl++) s += part[sl * 16384 + i];
  s += b1[i & 255];
  out1[i] = s > 0.0f ? s : 0.0f;
}

// ======================================================================
// Tiny head: relu(out1@W2+b2) @ W3 + b3 -> softmax. <0.1% of FLOPs, VALU.
// ======================================================================
__global__ __launch_bounds__(256) void head(const float* __restrict__ out1,
                                            const float* __restrict__ W2,
                                            const float* __restrict__ b2,
                                            const float* __restrict__ W3,
                                            const float* __restrict__ b3,
                                            float* __restrict__ out) {
  __shared__ float o2[64 * 32];
  int tid = threadIdx.x;
  for (int i = tid; i < 64 * 32; i += 256) {
    int m = i >> 5, n = i & 31;
    float s = b2[n];
    for (int k = 0; k < 256; k++) s += out1[m * 256 + k] * W2[k * 32 + n];
    o2[i] = s > 0.0f ? s : 0.0f;
  }
  __syncthreads();
  if (tid < 64) {
    float z0 = b3[0], z1 = b3[1];
#pragma unroll
    for (int j = 0; j < 32; j++) {
      float v = o2[tid * 32 + j];
      z0 += v * W3[j * 2];
      z1 += v * W3[j * 2 + 1];
    }
    float mx = fmaxf(z0, z1);
    float e0 = fast_exp(z0 - mx), e1 = fast_exp(z1 - mx);
    float inv = __builtin_amdgcn_rcpf(e0 + e1);
    out[tid * 2]     = e0 * inv;
    out[tid * 2 + 1] = e1 * inv;
  }
}

// ======================================================================
extern "C" void kernel_launch(void* const* d_in, const int* in_sizes, int n_in,
                              void* d_out, int out_size, void* d_ws, size_t ws_size,
                              hipStream_t stream) {
  const float* inp = (const float*)d_in[0];
  const float* Wn  = (const float*)d_in[1];
  const float* Wg  = (const float*)d_in[2];
  const float* Ug  = (const float*)d_in[3];
  const float* bg  = (const float*)d_in[4];
  const float* Wc  = (const float*)d_in[5];
  const float* Uc  = (const float*)d_in[6];
  const float* bc  = (const float*)d_in[7];
  const float* We  = (const float*)d_in[8];
  const float* be  = (const float*)d_in[9];
  const float* W1  = (const float*)d_in[10];
  const float* b1  = (const float*)d_in[11];
  const float* W2  = (const float*)d_in[12];
  const float* b2  = (const float*)d_in[13];
  const float* W3  = (const float*)d_in[14];
  const float* b3  = (const float*)d_in[15];
  (void)in_sizes; (void)n_in; (void)out_size; (void)ws_size;

  char* ws = (char*)d_ws;
  _Float16* rw   = (_Float16*)(ws + WS_RW_OFF);
  _Float16* Hbuf = (_Float16*)(ws + WS_H_OFF);
  _Float16* w1b  = (_Float16*)(ws + WS_W1_OFF);
  float*    part = (float*)(ws + WS_PART_OFF);
  float*    out1 = (float*)(ws + WS_OUT1_OFF);
  float*    out  = (float*)d_out;

  prep_rw<<<8, 256, 0, stream>>>(Wn, Wg, Ug, Wc, Uc, We, rw);
  prep_w1<<<KTOT, 256, 0, stream>>>(W1, w1b);              // 65536 blocks
  ugrnn_steps<<<TILES / 8, 256, 0, stream>>>(inp, rw, bg, bc, be, Hbuf);
  gemm1<<<(KSLICES * 64) / 8, 256, 0, stream>>>(Hbuf, w1b, part);
  reduce1<<<64, 256, 0, stream>>>(part, b1, out1);
  head<<<1, 256, 0, stream>>>(out1, W2, b2, W3, b3, out);
}
